// STCN_81484119540555
// MI455X (gfx1250) — matrix operand order repaired and checked by reading, not compiled, not run on hardware
//
#include <hip/hip_runtime.h>
#include <hip/hip_bf16.h>

typedef __attribute__((ext_vector_type(16))) __bf16 v16bf;
typedef __attribute__((ext_vector_type(8)))  float  v8f;
typedef __attribute__((ext_vector_type(8)))  int    v8i;

#define AS1 __attribute__((address_space(1)))
#define AS3 __attribute__((address_space(3)))

#if defined(__has_builtin)
# if __has_builtin(__builtin_amdgcn_global_load_async_to_lds_b32)
#  define HAVE_ASYNC_COPY 1
# endif
# if __has_builtin(__builtin_amdgcn_s_wait_asynccnt)
#  define HAVE_WAIT_ASYNC 1
# endif
#endif

#define B_   4
#define CK_  64
#define CV_  512
#define HW_  1620
#define THW_ 12960
#define KT   32      // t-block (K of PV gemm, M of logits gemm)
#define NQ   32      // q columns per workgroup
#define NBLK (THW_ / KT)   // 405

union FragBF { v8i i; v16bf h; };

__device__ __forceinline__ unsigned short f2bf(float f) {
    unsigned int u = __float_as_uint(f);
    unsigned int r = (u + 0x7FFFu + ((u >> 16) & 1u)) >> 16;  // RNE
    return (unsigned short)r;
}

__device__ __forceinline__ void wait_async_zero() {
#if defined(HAVE_ASYNC_COPY)
# if defined(HAVE_WAIT_ASYNC)
    __builtin_amdgcn_s_wait_asynccnt(0);
# else
    asm volatile("s_wait_asynccnt 0x0" ::: "memory");
# endif
#endif
}

// Stage one mv tile [512 rows][32 bf16] (row stride THW_ in global, 17 dwords in LDS).
__device__ __forceinline__ void stage_mv(const unsigned int* __restrict__ gmv,
                                         unsigned int* buf, int tid) {
#if defined(HAVE_ASYNC_COPY)
    #pragma unroll
    for (int r = 0; r < 32; ++r) {
        int idx = tid + r * 256;
        int cv  = idx >> 4;
        int d   = idx & 15;
        __builtin_amdgcn_global_load_async_to_lds_b32(
            (AS1 int*)(gmv + (size_t)cv * (THW_ / 2) + d),
            (AS3 int*)((AS3 unsigned int*)buf + cv * 17 + d),
            0, 0);
    }
#else
    for (int idx = tid; idx < CV_ * 16; idx += 256) {
        int cv = idx >> 4;
        int d  = idx & 15;
        buf[cv * 17 + d] = gmv[(size_t)cv * (THW_ / 2) + d];
    }
#endif
}

// ---- prep kernel 1: mk fp32 -> bf16, plus a_sq[b][t] = sum_c mk^2 ----
__global__ __launch_bounds__(256)
void prep_mk(const float* __restrict__ mk, unsigned short* __restrict__ mkb,
             float* __restrict__ asq) {
    int idx = blockIdx.x * blockDim.x + threadIdx.x;
    if (idx >= B_ * THW_) return;
    int b = idx / THW_;
    int t = idx - b * THW_;
    const float*    src = mk  + (size_t)b * CK_ * THW_ + t;
    unsigned short* dst = mkb + (size_t)b * CK_ * THW_ + t;
    float s = 0.f;
    #pragma unroll
    for (int c = 0; c < CK_; ++c) {
        float v = src[(size_t)c * THW_];
        s += v * v;
        dst[(size_t)c * THW_] = f2bf(v);
    }
    asq[idx] = s;
}

// ---- prep kernel 2: generic fp32 -> bf16 ----
__global__ __launch_bounds__(256)
void conv_bf16(const float* __restrict__ in, unsigned short* __restrict__ out, int n) {
    int idx = blockIdx.x * blockDim.x + threadIdx.x;
    if (idx < n) out[idx] = f2bf(in[idx]);
}

// ---- main fused attention kernel ----
__global__ __launch_bounds__(256)
void stcn_attn(const unsigned short* __restrict__ mkb,
               const unsigned short* __restrict__ qkb,
               const unsigned short* __restrict__ mvb,
               const float* __restrict__ asq,
               float* __restrict__ out) {
    // LDS (double-buffered mv/mk/asq); padded strides to dodge bank conflicts
    __shared__ unsigned int s_mvU[2][CV_ * 17];  // mv tile [512][32 bf16 + pad]
    __shared__ unsigned int s_mkU[2][KT * 34];   // mk tile [32 t][64 c + pad]
    __shared__ unsigned int s_qkU[NQ * 34];      // qk tile [32 q][64 c + pad]
    __shared__ unsigned int s_PU [NQ * 17];      // P tile  [32 q][32 t + pad] bf16
    __shared__ float s_asq[2][KT];
    __shared__ float s_tmax[2][NQ];
    __shared__ float s_tsum[2][NQ];
    __shared__ float s_M[NQ], s_S[NQ], s_scale[NQ], s_mnew[NQ];

    unsigned short* s_mkS = (unsigned short*)s_mkU;  // [buf*KT*68 + t*68 + c]
    unsigned short* s_qkS = (unsigned short*)s_qkU;

    const int b     = blockIdx.y;
    const int qBase = blockIdx.x * NQ;
    const int tid   = threadIdx.x;
    const int wave  = tid >> 5;
    const int lane  = tid & 31;
    const int ln    = lane & 15;   // N (or M) index within 16
    const int g     = lane >> 4;   // lane half

    const int mtile   = wave >> 1; // waves 0-3: logit tile row
    const int ntile_l = wave & 1;  // waves 0-3: logit tile col

    const unsigned int* gmvb = (const unsigned int*)(mvb + (size_t)b * CV_ * THW_);

    // ---- prologue: qk tile (whole WG lifetime), stats init, block-0 staging ----
    for (int idx = tid; idx < NQ * CK_; idx += 256) {
        int q = idx & (NQ - 1);
        int c = idx >> 5;
        int qg = qBase + q;
        unsigned short v = 0;
        if (qg < HW_) v = qkb[((size_t)b * CK_ + c) * HW_ + qg];
        s_qkS[q * 68 + c] = v;
    }
    if (tid < NQ) { s_M[tid] = -3.0e38f; s_S[tid] = 0.f; }
    stage_mv(gmvb, s_mvU[0], tid);
    for (int idx = tid; idx < KT * CK_; idx += 256) {
        int t = idx & (KT - 1);
        int c = idx >> 5;
        s_mkS[t * 68 + c] = mkb[((size_t)b * CK_ + c) * THW_ + t];
    }
    if (tid < KT) s_asq[0][tid] = asq[(size_t)b * THW_ + tid];

    v8f acc[4][2];
    #pragma unroll
    for (int mt = 0; mt < 4; ++mt)
        #pragma unroll
        for (int e = 0; e < 8; ++e) { acc[mt][0][e] = 0.f; acc[mt][1][e] = 0.f; }

    wait_async_zero();
    __syncthreads();

    for (int blk = 0; blk < NBLK; ++blk) {
        const int cur = blk & 1;
        const int nxt = cur ^ 1;
        const bool haveNext = (blk + 1) < NBLK;

        // ---- A: prefetch next block (async mv -> LDS; mk/asq -> registers) ----
        unsigned short mkreg[8];
        float asqreg = 0.f;
        if (haveNext) {
            const int tNext = (blk + 1) * KT;
            stage_mv(gmvb + tNext / 2, s_mvU[nxt], tid);
            #pragma unroll
            for (int r = 0; r < 8; ++r) {
                int idx = tid + r * 256;          // 2048 = KT*CK_
                int t = idx & (KT - 1);
                int c = idx >> 5;
                mkreg[r] = mkb[((size_t)b * CK_ + c) * THW_ + tNext + t];
            }
            if (tid < KT) asqreg = asq[(size_t)b * THW_ + (blk + 1) * KT + tid];
        }

        const unsigned int* mkB = s_mkU[cur];
        const unsigned int* mvB = s_mvU[cur];

        // ---- B: waves 0-3 compute 16x16 logit tiles via bf16 WMMA ----
        float lvals[8];
        if (wave < 4) {
            FragBF a0, a1, b0, b1;
            int rowA = mtile * 16 + ln;     // lane = M (t within block)
            int rowB = ntile_l * 16 + ln;   // lane = N (q within tile)
            #pragma unroll
            for (int v = 0; v < 8; ++v) {
                int offA = 8 * g + v;                            // A: K pairs
                int offB = 8 * g + v;                            // B: K pairs
                a0.i[v] = mkB[rowA * 34 + offA];
                a1.i[v] = mkB[rowA * 34 + 16 + offA];
                b0.i[v] = s_qkU[rowB * 34 + offB];
                b1.i[v] = s_qkU[rowB * 34 + 16 + offB];
            }
            v8f d;
            #pragma unroll
            for (int e = 0; e < 8; ++e) d[e] = 0.f;
            d = __builtin_amdgcn_wmma_f32_16x16x32_bf16(false, a0.h, false, b0.h,
                                                        (short)0, d, false, false);
            d = __builtin_amdgcn_wmma_f32_16x16x32_bf16(false, a1.h, false, b1.h,
                                                        (short)0, d, false, false);
            float lmax = -3.0e38f;
            #pragma unroll
            for (int v = 0; v < 8; ++v) {
                float aq = s_asq[cur][mtile * 16 + 8 * g + v];   // t = mtile*16+8g+v
                float l  = 0.25f * d[v] - 0.125f * aq;           // (2ab - asq)/sqrt(64)
                lvals[v] = l;
                lmax = fmaxf(lmax, l);
            }
            lmax = fmaxf(lmax, __shfl_xor(lmax, 16, 32));
            if (g == 0) s_tmax[mtile][ntile_l * 16 + ln] = lmax;
        }
        __syncthreads();

        // ---- C: online-softmax running max / rescale factors ----
        if (tid < NQ) {
            float bm   = fmaxf(s_tmax[0][tid], s_tmax[1][tid]);
            float mold = s_M[tid];
            float mnew = fmaxf(mold, bm);
            float sc   = __expf(mold - mnew);
            s_M[tid]     = mnew;
            s_S[tid]    *= sc;
            s_scale[tid] = sc;
            s_mnew[tid]  = mnew;
        }
        __syncthreads();

        // ---- D: waves 0-3 exp + write bf16 P; all waves rescale accumulators ----
        if (wave < 4) {
            int   q    = ntile_l * 16 + ln;
            float mnew = s_mnew[q];
            float lsum = 0.f;
            int base = q * 17 + mtile * 8 + 4 * g;
            #pragma unroll
            for (int v = 0; v < 8; v += 2) {
                float p0 = __expf(lvals[v]     - mnew);
                float p1 = __expf(lvals[v + 1] - mnew);
                lsum += p0 + p1;
                s_PU[base + (v >> 1)] =
                    (unsigned int)f2bf(p0) | ((unsigned int)f2bf(p1) << 16);
            }
            lsum += __shfl_xor(lsum, 16, 32);
            if (g == 0) s_tsum[mtile][q] = lsum;
        }
        {
            float sc0 = s_scale[ln];
            float sc1 = s_scale[16 + ln];
            // running max usually stable -> scale==1 -> skip the VALU burst
            if (!__all((sc0 == 1.0f) && (sc1 == 1.0f))) {
                #pragma unroll
                for (int mt = 0; mt < 4; ++mt)
                    #pragma unroll
                    for (int e = 0; e < 8; ++e) { acc[mt][0][e] *= sc0; acc[mt][1][e] *= sc1; }
            }
        }
        __syncthreads();

        // ---- E: fold block sums into running sum ----
        if (tid < NQ) s_S[tid] += s_tsum[0][tid] + s_tsum[1][tid];

        // ---- F: PV GEMM, 8 waves x (4 M-tiles x 2 N-tiles) bf16 WMMA ----
        FragBF bp0, bp1;
        #pragma unroll
        for (int v = 0; v < 8; ++v) {
            int offB = 8 * g + v;
            bp0.i[v] = s_PU[ln * 17 + offB];
            bp1.i[v] = s_PU[(16 + ln) * 17 + offB];
        }
        #pragma unroll
        for (int mt = 0; mt < 4; ++mt) {
            FragBF am;
            int row = wave * 64 + mt * 16 + ln;   // CV row
            #pragma unroll
            for (int v = 0; v < 8; ++v) {
                int offA = 8 * g + v;
                am.i[v] = mvB[row * 17 + offA];
            }
            acc[mt][0] = __builtin_amdgcn_wmma_f32_16x16x32_bf16(false, am.h, false, bp0.h,
                                                                 (short)0, acc[mt][0], false, false);
            acc[mt][1] = __builtin_amdgcn_wmma_f32_16x16x32_bf16(false, am.h, false, bp1.h,
                                                                 (short)0, acc[mt][1], false, false);
        }

        // ---- G: commit next block's mk/asq registers to LDS, wait async, fence ----
        if (haveNext) {
            #pragma unroll
            for (int r = 0; r < 8; ++r) {
                int idx = tid + r * 256;
                int t = idx & (KT - 1);
                int c = idx >> 5;
                s_mkS[nxt * (KT * 68) + t * 68 + c] = mkreg[r];
            }
            if (tid < KT) s_asq[nxt][tid] = asqreg;
        }
        wait_async_zero();
        __syncthreads();
    }

    // ---- epilogue: normalize by softmax sum and store fp32 output ----
    float r0 = 1.f / s_S[ln];
    float r1 = 1.f / s_S[16 + ln];
    int q0 = qBase + ln;
    int q1 = qBase + 16 + ln;
    #pragma unroll
    for (int mt = 0; mt < 4; ++mt) {
        #pragma unroll
        for (int e = 0; e < 8; ++e) {
            int cv = wave * 64 + mt * 16 + 8 * g + e;
            size_t rowoff = ((size_t)b * CV_ + cv) * HW_;
            if (q0 < HW_) out[rowoff + q0] = acc[mt][0][e] * r0;
            if (q1 < HW_) out[rowoff + q1] = acc[mt][1][e] * r1;
        }
    }
}

extern "C" void kernel_launch(void* const* d_in, const int* in_sizes, int n_in,
                              void* d_out, int out_size, void* d_ws, size_t ws_size,
                              hipStream_t stream) {
    const float* mk = (const float*)d_in[0];   // [4,64,8,30,54]
    const float* qk = (const float*)d_in[1];   // [4,64,30,54]
    const float* mv = (const float*)d_in[2];   // [4,512,8,30,54]
    float* out = (float*)d_out;                // [4,512,30,54]

    // workspace layout (~60.8 MB): asq | mk_bf16 | qk_bf16 | mv_bf16
    char* ws = (char*)d_ws;
    float*          asq = (float*)ws;                                   // 207360 B
    unsigned short* mkb = (unsigned short*)(ws + 207360);               // 6635520 B
    unsigned short* qkb = (unsigned short*)(ws + 207360 + 6635520);     // 829440 B
    unsigned short* mvb = (unsigned short*)(ws + 207360 + 6635520 + 829440); // 53084160 B
    (void)ws_size; (void)in_sizes; (void)n_in; (void)out_size;

    prep_mk<<<(B_ * THW_ + 255) / 256, 256, 0, stream>>>(mk, mkb, asq);
    conv_bf16<<<(B_ * CK_ * HW_ + 255) / 256, 256, 0, stream>>>(qk, qkb, B_ * CK_ * HW_);
    conv_bf16<<<(B_ * CV_ * THW_ + 255) / 256, 256, 0, stream>>>(mv, mvb, B_ * CV_ * THW_);

    dim3 grid((HW_ + NQ - 1) / NQ, B_);   // 51 x 4
    stcn_attn<<<grid, 256, 0, stream>>>(mkb, qkb, mvb, asq, out);
}
